// TinyGRU_42262478192913
// MI455X (gfx1250) — matrix operand order repaired and checked by reading, not compiled, not run on hardware
//
#include <hip/hip_runtime.h>
#include <hip/hip_bf16.h>

// ---- problem constants (from reference) ----
#define BB    2048
#define SS    512
#define EE    48      // embed dim
#define HH    48      // hidden dim
#define GG    144     // 3*H
#define VV    95      // vocab
#define VPAD  96
#define KPAD  64      // K padded for 16x16x32 f16 WMMA (2 chunks of 32)
#define BLOCK 256     // 8 wave32s
#define MTILE 16      // batch rows per workgroup

typedef _Float16 v8h  __attribute__((ext_vector_type(8)));
typedef _Float16 v16h __attribute__((ext_vector_type(16)));
typedef float    v8f  __attribute__((ext_vector_type(8)));

// ---- dynamic LDS layout (bytes, all 16B aligned) ----
#define OFF_WIH   0                      // 18 frags * 512 halfs = 18432 B (w_ih^T, B-frag swizzled)
#define OFF_WHH   18432                  // 18432 B
#define OFF_WOUT  36864                  // 12 frags * 512 halfs = 12288 B
#define OFF_XF    49152                  // x_f16 [16][64]  = 2048 B
#define OFF_HF    51200                  // h_f16 [16][64]  = 2048 B
#define OFF_H     53248                  // h f32 [16][48]  = 3072 B
#define OFF_XG    56320                  // xg f32 [16][144] = 9216 B
#define OFF_GH    65536                  // gh f32 [16][144] = 9216 B
#define OFF_BIH   74752                  // 144 f32 = 576 B
#define OFF_BHH   75328                  // 576 B
#define OFF_BOUT  75904                  // 96 f32 = 384 B
#define SMEM_BYTES 76288

// Fast device math: single v_rcp_f32 / hardware v_tanh_f32 (CDNA5 trans ops)
// instead of the IEEE-division fixup chain / ocml tanh expansion.
__device__ __forceinline__ float fast_sigmoid(float x) {
#if __has_builtin(__builtin_amdgcn_rcpf)
    return __builtin_amdgcn_rcpf(1.0f + __expf(-x));
#else
    return 1.0f / (1.0f + __expf(-x));
#endif
}
__device__ __forceinline__ float fast_tanh(float x) {
#if __has_builtin(__builtin_amdgcn_tanhf)
    return __builtin_amdgcn_tanhf(x);
#else
    return tanhf(x);
#endif
}

// Load a lane's 16-half A fragment (16x32 f16 A-matrix layout, row-major LDS
// [16][KPAD]). Each lane's data is two contiguous 16-byte chunks.
__device__ __forceinline__ v16h load_fragA(const _Float16* A, int lane, int k0) {
    int m = lane & 15, half = lane >> 4;
    const _Float16* base = A + m * KPAD + k0 + half * 8;
    v8h lo = *(const v8h*)(base);        // K = k0 + half*8 + 0..7
    v8h hi = *(const v8h*)(base + 16);   // K = k0 + 16 + half*8 + 0..7
    return __builtin_shufflevector(lo, hi, 0,1,2,3,4,5,6,7,8,9,10,11,12,13,14,15);
}

// Load a lane's 16-half pre-swizzled B fragment (fully contiguous, 32 bytes).
__device__ __forceinline__ v16h load_fragB(const _Float16* frags, int fragId, int lane) {
    const _Float16* base = frags + (fragId << 9) + (lane << 4);
    v8h lo = *(const v8h*)(base);
    v8h hi = *(const v8h*)(base + 8);
    return __builtin_shufflevector(lo, hi, 0,1,2,3,4,5,6,7,8,9,10,11,12,13,14,15);
}

// Fill one element of a swizzled B-fragment store from row-major W[N][K].
// B(k,n) = W[n][k]. Fragment element (lane, e): k = kc*32 + (e>>3)*16 + (lane>>4)*8 + (e&7),
// n = nt*16 + (lane&15). Zero-pads K 48..63 and N >= Nvalid.
__device__ __forceinline__ void fill_fragB(_Float16* dst, const float* W,
                                           int idx, int Nvalid, int K) {
    int f = idx >> 9;            // fragment id = nt*2 + kc
    int r = idx & 511;
    int lane = r >> 4, e = r & 15;
    int kc = f & 1, nt = f >> 1;
    int k = kc * 32 + ((e >> 3) << 4) + ((lane >> 4) << 3) + (e & 7);
    int n = (nt << 4) + (lane & 15);
    float v = (k < K && n < Nvalid) ? W[n * K + k] : 0.0f;
    dst[idx] = (_Float16)v;
}

__global__ void __launch_bounds__(BLOCK)
tiny_gru_fused(const int* __restrict__ xp, const float* __restrict__ emb,
               const float* __restrict__ w_ih, const float* __restrict__ w_hh,
               const float* __restrict__ b_ih, const float* __restrict__ b_hh,
               const float* __restrict__ w_out, const float* __restrict__ b_out,
               float* __restrict__ out) {
    extern __shared__ __align__(16) char smem[];
    _Float16* wihS  = (_Float16*)(smem + OFF_WIH);
    _Float16* whhS  = (_Float16*)(smem + OFF_WHH);
    _Float16* woutS = (_Float16*)(smem + OFF_WOUT);
    _Float16* xf    = (_Float16*)(smem + OFF_XF);
    _Float16* hf    = (_Float16*)(smem + OFF_HF);
    float*    h     = (float*)(smem + OFF_H);
    float*    xg    = (float*)(smem + OFF_XG);
    float*    gh    = (float*)(smem + OFF_GH);
    float*    bih   = (float*)(smem + OFF_BIH);
    float*    bhh   = (float*)(smem + OFF_BHH);
    float*    bout  = (float*)(smem + OFF_BOUT);

    const int tid  = threadIdx.x;
    const int wave = tid >> 5;
    const int lane = tid & 31;
    const int b0   = blockIdx.x * MTILE;
    const size_t HID_OFF = (size_t)BB * SS * VV;

    // ---- init: swizzle weights to f16 B-fragments, stage biases, zero state ----
    for (int i = tid; i < 18 * 512; i += BLOCK) fill_fragB(wihS,  w_ih,  i, GG, EE);
    for (int i = tid; i < 18 * 512; i += BLOCK) fill_fragB(whhS,  w_hh,  i, GG, HH);
    for (int i = tid; i < 12 * 512; i += BLOCK) fill_fragB(woutS, w_out, i, VV, HH);
    for (int i = tid; i < GG; i += BLOCK) { bih[i] = b_ih[i]; bhh[i] = b_hh[i]; }
    for (int i = tid; i < VPAD; i += BLOCK) bout[i] = (i < VV) ? b_out[i] : 0.0f;
    for (int i = tid; i < MTILE * KPAD; i += BLOCK) { xf[i] = (_Float16)0.0f; hf[i] = (_Float16)0.0f; }
    for (int i = tid; i < MTILE * HH; i += BLOCK) h[i] = 0.0f;
    // gather timestep 0 embeddings
    for (int i = tid; i < MTILE * EE; i += BLOCK) {
        int m = i / EE, e = i - m * EE;
        int tok = xp[(size_t)(b0 + m) * SS + 0];
        xf[m * KPAD + e] = (_Float16)emb[tok * EE + e];
    }
    __syncthreads();

    // ---- sequential GRU over S timesteps (3 barriers per step) ----
    for (int t = 0; t < SS; ++t) {
        // Phase B: 18 gate tiles: xg = x @ w_ih^T  and  gh = h @ w_hh^T
        for (int tt = wave; tt < 18; tt += 8) {
            const bool isX = tt < 9;
            const int nt = isX ? tt : tt - 9;
            const _Float16* A  = isX ? xf   : hf;
            const _Float16* Bw = isX ? wihS : whhS;
            float*          C  = isX ? xg   : gh;
            v8f acc = {0.f, 0.f, 0.f, 0.f, 0.f, 0.f, 0.f, 0.f};
            #pragma unroll
            for (int kc = 0; kc < 2; ++kc) {
                v16h a = load_fragA(A, lane, kc * 32);
                v16h b = load_fragB(Bw, nt * 2 + kc, lane);
                acc = __builtin_amdgcn_wmma_f32_16x16x32_f16(
                          false, a, false, b, (short)0, acc, false, false);
            }
            const int n  = (nt << 4) + (lane & 15);
            const int mb = (lane >> 4) << 3;
            #pragma unroll
            for (int v = 0; v < 8; ++v) C[(mb + v) * GG + n] = acc[v];
        }
        __syncthreads();

        // Phase C: gate nonlinearity + hidden update (f32), refresh hf (f16)
        for (int i = tid; i < MTILE * HH; i += BLOCK) {
            int m = i / HH, j = i - m * HH;
            float xr = xg[m * GG + j]            + bih[j];
            float xz = xg[m * GG + HH + j]       + bih[HH + j];
            float xn = xg[m * GG + 2 * HH + j]   + bih[2 * HH + j];
            float hr = gh[m * GG + j]            + bhh[j];
            float hz = gh[m * GG + HH + j]       + bhh[HH + j];
            float hn = gh[m * GG + 2 * HH + j]   + bhh[2 * HH + j];
            float r  = fast_sigmoid(xr + hr);
            float z  = fast_sigmoid(xz + hz);
            float ng = fast_tanh(xn + r * hn);
            float hv = (1.0f - z) * ng + z * h[i];
            h[i] = hv;
            hf[m * KPAD + j] = (_Float16)hv;
        }
        __syncthreads();

        // Phase D (wave-specialized, overlapped):
        //   waves 0..5: logits tile = h_new @ w_out^T + b_out -> global
        //   waves 6..7: gather next timestep's embeddings into xf
        if (wave < 6) {
            const int tt = wave;
            v8f acc = {0.f, 0.f, 0.f, 0.f, 0.f, 0.f, 0.f, 0.f};
            #pragma unroll
            for (int kc = 0; kc < 2; ++kc) {
                v16h a = load_fragA(hf, lane, kc * 32);
                v16h b = load_fragB(woutS, tt * 2 + kc, lane);
                acc = __builtin_amdgcn_wmma_f32_16x16x32_f16(
                          false, a, false, b, (short)0, acc, false, false);
            }
            const int n  = (tt << 4) + (lane & 15);
            const int mb = (lane >> 4) << 3;
            if (n < VV) {
                const float bb = bout[n];
                #pragma unroll
                for (int v = 0; v < 8; ++v) {
                    int m = mb + v;
                    out[((size_t)(b0 + m) * SS + t) * VV + n] = acc[v] + bb;
                }
            }
        } else if (t + 1 < SS) {
            for (int i = tid - 6 * 32; i < MTILE * EE; i += 64) {
                int m = i / EE, e = i - m * EE;
                int tok = xp[(size_t)(b0 + m) * SS + (t + 1)];
                xf[m * KPAD + e] = (_Float16)emb[tok * EE + e];
                if (e == 0 && t + 2 < SS)
                    __builtin_prefetch(&xp[(size_t)(b0 + m) * SS + t + 2], 0, 0);
            }
        }
        __syncthreads();
    }

    // ---- epilogue: final hidden state [1, B, H] appended after logits ----
    for (int i = tid; i < MTILE * HH; i += BLOCK) {
        int m = i / HH, j = i - m * HH;
        out[HID_OFF + (size_t)(b0 + m) * HH + j] = h[i];
    }
}

extern "C" void kernel_launch(void* const* d_in, const int* in_sizes, int n_in,
                              void* d_out, int out_size, void* d_ws, size_t ws_size,
                              hipStream_t stream) {
    (void)in_sizes; (void)n_in; (void)out_size; (void)d_ws; (void)ws_size;
    const int*   xp    = (const int*)d_in[0];
    const float* emb   = (const float*)d_in[1];
    const float* w_ih  = (const float*)d_in[2];
    const float* w_hh  = (const float*)d_in[3];
    const float* b_ih  = (const float*)d_in[4];
    const float* b_hh  = (const float*)d_in[5];
    const float* w_out = (const float*)d_in[6];
    const float* b_out = (const float*)d_in[7];
    float*       out   = (float*)d_out;

    // Allow >64KB dynamic LDS (CDNA5 WGP has 320KB).
    hipFuncSetAttribute((const void*)tiny_gru_fused,
                        hipFuncAttributeMaxDynamicSharedMemorySize, SMEM_BYTES);

    dim3 grid(BB / MTILE);   // 128 workgroups, one 16-row batch tile each
    dim3 block(BLOCK);       // 8 wave32s
    tiny_gru_fused<<<grid, block, SMEM_BYTES, stream>>>(
        xp, emb, w_ih, w_hh, b_ih, b_hh, w_out, b_out, out);
}
